// RNN_Skills_Model_78082505441617
// MI455X (gfx1250) — matrix operand order repaired and checked by reading, not compiled, not run on hardware
//
#include <hip/hip_runtime.h>
#include <hip/hip_bf16.h>

// ---------------- problem constants ----------------
#define QN   512      // questions (time steps)
#define NN   1000     // students / hidden size
#define HH   1024     // input embedding size
#define CC   128      // concepts
#define LL   8        // lstm layers
#define G4   4000     // 4*N gate rows
#define KP   1024     // padded K (1000 -> 1024)
#define LAMB 10.0f
#define GUESS 0.25f

typedef __attribute__((ext_vector_type(16))) __bf16 v16bf;
typedef __attribute__((ext_vector_type(8)))  float  v8f;

// ---------------- helpers ----------------
__device__ __forceinline__ unsigned short f2bf(float f) {
    unsigned u = __float_as_uint(f);
    unsigned r = u + 0x7FFFu + ((u >> 16) & 1u);   // RNE
    return (unsigned short)(r >> 16);
}
__device__ __forceinline__ float bflo(unsigned v) { return __uint_as_float(v << 16); }
__device__ __forceinline__ float bfhi(unsigned v) { return __uint_as_float(v & 0xFFFF0000u); }
__device__ __forceinline__ float sigf(float x) { return 1.0f / (1.0f + __expf(-x)); }

// ---------------- fp32 -> bf16 convert with K padding ----------------
__global__ void cvt_pad_bf16(const float* __restrict__ src, unsigned short* __restrict__ dst,
                             int rows, int cols, int colsPad) {
    long idx = (long)blockIdx.x * 256 + threadIdx.x;
    long total = (long)rows * colsPad;
    if (idx >= total) return;
    int r = (int)(idx / colsPad), k = (int)(idx % colsPad);
    float v = (k < cols) ? src[(long)r * cols + k] : 0.0f;
    dst[idx] = f2bf(v);
}

__global__ void add_vec(const float* __restrict__ a, const float* __restrict__ b,
                        float* __restrict__ o, int n) {
    int i = blockIdx.x * 256 + threadIdx.x;
    if (i < n) o[i] = a[i] + b[i];
}

// ---------------- WMMA tile: C(16x16) = A(16xK) * B(16xK)^T, bf16 inputs ----------------
// A row-major [*, lda], B row-major [*, ldb], K multiple of 32.
// A-frag per ISA: lane m=lane&15, hi=lane>>4, halves K+{8*hi..} and K+{16+8*hi..}
// B-frag per ISA: lane n=lane&15, halves K+{8*hi..} and K+{16+8*hi..}
__device__ __forceinline__ v8f wmma_tile(const unsigned short* __restrict__ A,
                                         const unsigned short* __restrict__ B,
                                         int m0, int n0, int K, int lda, int ldb, int lane) {
    int mr = lane & 15, hi = lane >> 4;
    const unsigned short* arow = A + (size_t)(m0 + mr) * lda;
    const unsigned short* brow = B + (size_t)(n0 + mr) * ldb;
    v8f acc = {};
    for (int k = 0; k < K; k += 32) {
        union { uint4 q[2]; v16bf v; } af, bb;
        af.q[0] = *(const uint4*)(arow + k + 8 * hi);
        af.q[1] = *(const uint4*)(arow + k + 16 + 8 * hi);
        bb.q[0] = *(const uint4*)(brow + k + 8 * hi);
        bb.q[1] = *(const uint4*)(brow + k + 16 + 8 * hi);
        acc = __builtin_amdgcn_wmma_f32_16x16x32_bf16(false, af.v, false, bb.v,
                                                      (short)0, acc, false, false);
    }
    return acc;
}

// ---------------- GEMM 1: Xpre[q][h] = sum_n W_inp[h][n]*inputs[q][n] + b_inp[h], bf16 out ----
__global__ void gemm_xpre(const unsigned short* __restrict__ A,   // W_inp bf16 [1024][1024]
                          const unsigned short* __restrict__ B,   // inputs bf16 [512][1024]
                          const float* __restrict__ b_inp,
                          unsigned short* __restrict__ Xpre) {    // [512][1024] bf16
    int lane = threadIdx.x & 31, wv = threadIdx.x >> 5;
    int tile = blockIdx.x * 8 + wv;                    // 64*32 = 2048 tiles
    const int Qt = QN / 16;
    int m0 = (tile / Qt) * 16, q0 = (tile % Qt) * 16;
    v8f acc = wmma_tile(A, B, m0, q0, KP, KP, KP, lane);
    int n = lane & 15, hi = lane >> 4;
    int mb = m0 + 8 * hi;
    union { v8f v; float f[8]; } r; r.v = acc;
    union { unsigned short s[8]; uint4 q; } o;
#pragma unroll
    for (int v = 0; v < 8; ++v) o.s[v] = f2bf(r.f[v] + b_inp[mb + v]);
    *(uint4*)(Xpre + (size_t)(q0 + n) * KP + mb) = o.q;
}

// ---------------- GEMM 2: G0pre[q][m] = sum_h W_ih0[m][h]*Xpre[q][h] + bih0[m]+bhh0[m] ------
__global__ void gemm_g0(const unsigned short* __restrict__ A,    // W_ih0 bf16 [4000][1024]
                        const unsigned short* __restrict__ B,    // Xpre bf16 [512][1024]
                        const float* __restrict__ bih0, const float* __restrict__ bhh0,
                        float* __restrict__ G0pre) {             // [512][4000] f32
    int lane = threadIdx.x & 31, wv = threadIdx.x >> 5;
    int tile = blockIdx.x * 8 + wv;                    // 250*32 = 8000 tiles
    const int Qt = QN / 16;
    int m0 = (tile / Qt) * 16, q0 = (tile % Qt) * 16;
    v8f acc = wmma_tile(A, B, m0, q0, KP, KP, KP, lane);
    int n = lane & 15, hi = lane >> 4;
    int mb = m0 + 8 * hi;
    union { v8f v; float f[8]; float4 q[2]; } r; r.v = acc;
#pragma unroll
    for (int v = 0; v < 8; ++v) r.f[v] += bih0[mb + v] + bhh0[mb + v];
    float* orow = G0pre + (size_t)(q0 + n) * G4 + mb;
    *(float4*)(orow)     = r.q[0];
    *(float4*)(orow + 4) = r.q[1];
}

// ---------------- LSTM layer step: one launch per (layer, t) -------------------------------
// Block = 256 thr (8 waves), grid = 125 blocks. Wave w handles unit u = blk*8+w, computing
// gate rows g*1000+u for g=0..3 (dot over bf16 rows vs LDS-staged vectors), then the cell.
__global__ void lstm_layer_step(const unsigned short* __restrict__ Wih,  // [4000][1024] or unused
                                const unsigned short* __restrict__ Whh,  // [4000][1024]
                                const float* __restrict__ addvec,        // G0pre row (layer0) or bias[4000]
                                const float* __restrict__ xin,           // h_{l-1} new, padded 1024
                                const float* __restrict__ hin,           // h_l old, padded 1024
                                float* __restrict__ hout,                // h_l new
                                float* __restrict__ cst,                 // c_l in/out
                                int hasIh) {
    __shared__ float xs[KP];
    __shared__ float hs[KP];
    __shared__ float gsum[8][4];
    const int tid = threadIdx.x, lane = tid & 31, wv = tid >> 5;
    for (int i = tid; i < KP; i += 256) {
        hs[i] = hin[i];
        if (hasIh) xs[i] = xin[i];
    }
    __syncthreads();
    const int u = blockIdx.x * 8 + wv;                 // hidden unit 0..999
    for (int g = 0; g < 4; ++g) {
        int row = g * NN + u;
        float acc = 0.0f;
        const unsigned short* wr = Whh + (size_t)row * KP;
#pragma unroll
        for (int j = 0; j < 4; ++j) {
            int k = (j * 32 + lane) * 8;
            uint4 w4 = *(const uint4*)(wr + k);
            acc += bflo(w4.x) * hs[k + 0] + bfhi(w4.x) * hs[k + 1]
                 + bflo(w4.y) * hs[k + 2] + bfhi(w4.y) * hs[k + 3]
                 + bflo(w4.z) * hs[k + 4] + bfhi(w4.z) * hs[k + 5]
                 + bflo(w4.w) * hs[k + 6] + bfhi(w4.w) * hs[k + 7];
        }
        if (hasIh) {
            const unsigned short* wi = Wih + (size_t)row * KP;
#pragma unroll
            for (int j = 0; j < 4; ++j) {
                int k = (j * 32 + lane) * 8;
                uint4 w4 = *(const uint4*)(wi + k);
                acc += bflo(w4.x) * xs[k + 0] + bfhi(w4.x) * xs[k + 1]
                     + bflo(w4.y) * xs[k + 2] + bfhi(w4.y) * xs[k + 3]
                     + bflo(w4.z) * xs[k + 4] + bfhi(w4.z) * xs[k + 5]
                     + bflo(w4.w) * xs[k + 6] + bfhi(w4.w) * xs[k + 7];
            }
        }
#pragma unroll
        for (int off = 16; off; off >>= 1) acc += __shfl_xor(acc, off);
        if (lane == 0) gsum[wv][g] = acc + addvec[row];
    }
    __syncthreads();
    if (lane == 0) {                                   // one lane per wave finalizes its unit
        float gi = sigf(gsum[wv][0]);
        float gf = sigf(gsum[wv][1]);
        float gg = tanhf(gsum[wv][2]);
        float go = sigf(gsum[wv][3]);
        float c  = gf * cst[u] + gi * gg;
        cst[u]  = c;
        hout[u] = go * tanhf(c);
    }
}

// ---------------- skills head per step: single block, 1024 threads -------------------------
__global__ void skills_step(const float* __restrict__ h7,      // padded 1024, first 1000 valid
                            const float* __restrict__ W_lin, const float* __restrict__ b_lin,
                            const int* __restrict__ concept_col, const float* __restrict__ D_nz,
                            int t, float* __restrict__ out, float* __restrict__ ssum) {
    __shared__ float sh[KP];
    __shared__ float smn[32], smx[32];
    const int tid = threadIdx.x, lane = tid & 31, wv = tid >> 5;
    for (int i = tid; i < KP; i += 1024) sh[i] = h7[i];
    __syncthreads();
    float mn = 3.4e38f, mx = -3.4e38f;
    for (int idx = tid; idx < NN * CC; idx += 1024) {
        int n = idx >> 7, c = idx & 127;
        float v = sh[n] * W_lin[c] + b_lin[c];
        mn = fminf(mn, v); mx = fmaxf(mx, v);
    }
#pragma unroll
    for (int off = 16; off; off >>= 1) {
        mn = fminf(mn, __shfl_xor(mn, off));
        mx = fmaxf(mx, __shfl_xor(mx, off));
    }
    if (lane == 0) { smn[wv] = mn; smx[wv] = mx; }
    __syncthreads();
    if (tid == 0) {
        float a = smn[0], b = smx[0];
        for (int w = 1; w < 32; ++w) { a = fminf(a, smn[w]); b = fmaxf(b, smx[w]); }
        smn[0] = a; smx[0] = b;
    }
    __syncthreads();
    mn = smn[0]; mx = smx[0];
    float inv = 1.0f / (mx - mn);
    for (int idx = tid; idx < NN * CC; idx += 1024) {
        int n = idx >> 7, c = idx & 127;
        float s = (sh[n] * W_lin[c] + b_lin[c] - mn) * inv;
        ssum[idx] += s * (1.0f / (float)QN);
    }
    int cc = concept_col[t];
    float d = D_nz[t], wl = W_lin[cc], bl = b_lin[cc];
    for (int n = tid; n < NN; n += 1024) {
        float s = (sh[n] * wl + bl - mn) * inv;
        out[(size_t)t * NN + n] = fmaxf(GUESS, 1.0f - __expf(-LAMB * (s - d)));
    }
}

// ---------------- host launch ----------------
extern "C" void kernel_launch(void* const* d_in, const int* in_sizes, int n_in,
                              void* d_out, int out_size, void* d_ws, size_t ws_size,
                              hipStream_t stream) {
    (void)in_sizes; (void)n_in; (void)out_size; (void)ws_size;
    const float* inputs      = (const float*)d_in[0];
    const int*   concept_col = (const int*)  d_in[1];
    const float* W_inp_f     = (const float*)d_in[2];
    const float* b_inp       = (const float*)d_in[3];
    const float* W_ih0_f     = (const float*)d_in[4];
    const float* W_hh0_f     = (const float*)d_in[5];
    const float* b_ih0       = (const float*)d_in[6];
    const float* b_hh0       = (const float*)d_in[7];
    const float* W_ih_f      = (const float*)d_in[8];
    const float* W_hh_f      = (const float*)d_in[9];
    const float* b_ih        = (const float*)d_in[10];
    const float* b_hh        = (const float*)d_in[11];
    const float* W_lin       = (const float*)d_in[12];
    const float* b_lin       = (const float*)d_in[13];
    const float* D_nz        = (const float*)d_in[14];

    // workspace carve-up (256B aligned)
    char* base = (char*)d_ws;
    size_t off = 0;
    auto carve = [&](size_t bytes) { char* p = base + off; off = (off + bytes + 255) & ~(size_t)255; return p; };
    unsigned short* wInp = (unsigned short*)carve((size_t)HH * KP * 2);          // 2 MB
    unsigned short* bin  = (unsigned short*)carve((size_t)QN * KP * 2);          // 1 MB
    unsigned short* wIh0 = (unsigned short*)carve((size_t)G4 * KP * 2);          // 8 MB
    unsigned short* wHh0 = (unsigned short*)carve((size_t)G4 * KP * 2);          // 8 MB
    unsigned short* wIh  = (unsigned short*)carve((size_t)7 * G4 * KP * 2);      // 56 MB
    unsigned short* wHh  = (unsigned short*)carve((size_t)7 * G4 * KP * 2);      // 56 MB
    unsigned short* xpre = (unsigned short*)carve((size_t)QN * KP * 2);          // 1 MB
    float* g0pre = (float*)carve((size_t)QN * G4 * 4);                           // 8 MB
    float* biasL = (float*)carve((size_t)7 * G4 * 4);
    float* Hbuf  = (float*)carve((size_t)2 * LL * KP * 4);                       // ping-pong h
    float* Cbuf  = (float*)carve((size_t)LL * KP * 4);

    // init state + output accumulators
    hipMemsetAsync(Hbuf, 0, (size_t)2 * LL * KP * 4, stream);
    hipMemsetAsync(Cbuf, 0, (size_t)LL * KP * 4, stream);
    hipMemsetAsync(d_out, 0, (size_t)(QN * NN + NN * CC) * 4, stream);

    // bf16 conversions (K padded 1000 -> 1024 with zeros)
    auto grid = [](long total) { return (unsigned)((total + 255) / 256); };
    cvt_pad_bf16<<<grid((long)HH * KP), 256, 0, stream>>>(W_inp_f, wInp, HH, NN, KP);
    cvt_pad_bf16<<<grid((long)QN * KP), 256, 0, stream>>>(inputs, bin, QN, NN, KP);
    cvt_pad_bf16<<<grid((long)G4 * KP), 256, 0, stream>>>(W_ih0_f, wIh0, G4, HH, KP);
    cvt_pad_bf16<<<grid((long)G4 * KP), 256, 0, stream>>>(W_hh0_f, wHh0, G4, NN, KP);
    cvt_pad_bf16<<<grid((long)7 * G4 * KP), 256, 0, stream>>>(W_ih_f, wIh, 7 * G4, NN, KP);
    cvt_pad_bf16<<<grid((long)7 * G4 * KP), 256, 0, stream>>>(W_hh_f, wHh, 7 * G4, NN, KP);
    add_vec<<<grid(7 * G4), 256, 0, stream>>>(b_ih, b_hh, biasL, 7 * G4);

    // hoisted input-path GEMMs (WMMA bf16)
    gemm_xpre<<<(HH / 16) * (QN / 16) / 8, 256, 0, stream>>>(wInp, bin, b_inp, xpre);
    gemm_g0<<<(G4 / 16) * (QN / 16) / 8, 256, 0, stream>>>(wIh0, xpre, b_ih0, b_hh0, g0pre);

    // sequential recurrence
    float* outp = (float*)d_out;
    float* ssum = (float*)d_out + (size_t)QN * NN;
    for (int t = 0; t < QN; ++t) {
        int p = t & 1;
        float* hP = Hbuf + (size_t)p * LL * KP;        // previous step h
        float* hN = Hbuf + (size_t)(1 - p) * LL * KP;  // current step h
        lstm_layer_step<<<125, 256, 0, stream>>>(
            nullptr, wHh0, g0pre + (size_t)t * G4, nullptr,
            hP, hN, Cbuf, 0);
        for (int l = 1; l < LL; ++l) {
            lstm_layer_step<<<125, 256, 0, stream>>>(
                wIh + (size_t)(l - 1) * G4 * KP, wHh + (size_t)(l - 1) * G4 * KP,
                biasL + (size_t)(l - 1) * G4,
                hN + (size_t)(l - 1) * KP, hP + (size_t)l * KP,
                hN + (size_t)l * KP, Cbuf + (size_t)l * KP, 1);
        }
        skills_step<<<1, 1024, 0, stream>>>(hN + (size_t)(LL - 1) * KP,
                                            W_lin, b_lin, concept_col, D_nz, t, outp, ssum);
    }
}
